// PerformerModel_85358180041338
// MI455X (gfx1250) — matrix operand order repaired and checked by reading, not compiled, not run on hardware
//
#include <hip/hip_runtime.h>
#include <hip/hip_bf16.h>
#include <math.h>
#include <stdint.h>

// ---------------------------------------------------------------------------
// CDNA5 (gfx1250) WMMA helpers. wave32 only.
// ---------------------------------------------------------------------------
typedef __attribute__((ext_vector_type(16))) _Float16 v16h;
typedef __attribute__((ext_vector_type(8)))  float    v8f;
typedef __attribute__((ext_vector_type(4)))  unsigned int u32x4;
typedef __attribute__((ext_vector_type(8)))  int i32x8;
typedef __attribute__((ext_vector_type(4)))  int i32x4;
typedef __attribute__((ext_vector_type(4)))  int v4i;

// Pointer types matching the async-to-LDS builtin signature (v4i as1* / as3*)
typedef __attribute__((address_space(1))) v4i* gv4i_p;
typedef __attribute__((address_space(3))) v4i* lv4i_p;

#if __has_builtin(__builtin_amdgcn_global_load_async_to_lds_b128)
#define HAS_ASYNC_LDS 1
#else
#define HAS_ASYNC_LDS 0
#endif
#if __has_builtin(__builtin_amdgcn_tensor_load_to_lds)
#define HAS_TDM 1
#else
#define HAS_TDM 0
#endif

__device__ __forceinline__ v8f wmma_f(v16h a, v16h b, v8f c) {
  // D = A(16x32 f16) * B(32x16 f16) + C(16x16 f32)
  return __builtin_amdgcn_wmma_f32_16x16x32_f16(
      /*neg_a=*/false, a, /*neg_b=*/false, b,
      /*c_mod=*/(short)0, c, /*reuse_a=*/false, /*reuse_b=*/false);
}

// A tile (M=16 x K=32), src row-major [M][K] with leading dim ld (halves).
// Lane l<16: row M=l, K=0..15; lane l>=16: row M=l-16, K=16..31.
__device__ __forceinline__ v16h load_a_f16(const _Float16* __restrict__ src, int ld) {
  int lane = threadIdx.x & 31;
  int r = lane & 15;
  int kb = (lane & 16) ? 16 : 0;
  const _Float16* p = src + (size_t)r * ld + kb;
  v16h a;
#pragma unroll
  for (int j = 0; j < 8; ++j) a[j] = p[j];
#pragma unroll
  for (int j = 0; j < 8; ++j) a[8 + j] = p[8 + j];
  return a;
}

// B tile (K=32 x N=16) from an [N][K] row-major array (weights stored out-major),
// src points at row n0. Lane l<16: N=l, K=0..15; lane l>=16: N=l-16, K=16..31.
__device__ __forceinline__ v16h load_b_nk_f16(const _Float16* __restrict__ src, int ld) {
  int lane = threadIdx.x & 31;
  int n = lane & 15;
  int kb = (lane & 16) ? 16 : 0;
  const _Float16* p = src + (size_t)n * ld + kb;
  v16h b;
#pragma unroll
  for (int j = 0; j < 16; ++j) b[j] = p[j];
  return b;
}

// B tile (K=32 x N=16) from a [K][N] row-major array, src points at (k0, n0).
__device__ __forceinline__ v16h load_b_kn_f16(const _Float16* __restrict__ src, int ld) {
  int lane = threadIdx.x & 31;
  int n = lane & 15;
  int kb = (lane & 16) ? 16 : 0;
  v16h b;
#pragma unroll
  for (int j = 0; j < 16; ++j) b[j] = src[(size_t)(kb + j) * ld + n];
  return b;
}

// ---------------------------------------------------------------------------
// Async chunk staging: Qc/Kc via GLOBAL_LOAD_ASYNC_TO_LDS_B128 (ASYNCcnt),
// Vc via the Tensor Data Mover (2D D# tile, TENSORcnt), with plain-copy
// fallbacks so compilation never regresses.
// ---------------------------------------------------------------------------
__device__ __forceinline__ void copy_chunk(const _Float16* __restrict__ qsrc,
                                           const _Float16* __restrict__ ksrc,
                                           const _Float16* __restrict__ vsrc,
                                           _Float16* Qd, _Float16* Kd,
                                           _Float16* Vd, int tid) {
#if HAS_ASYNC_LDS
  for (int i = tid; i < 2048; i += 256) {
    __builtin_amdgcn_global_load_async_to_lds_b128(
        (gv4i_p)(uintptr_t)(qsrc + (size_t)i * 8),
        (lv4i_p)(uint32_t)(uintptr_t)(Qd + (size_t)i * 8), 0, 0);
    __builtin_amdgcn_global_load_async_to_lds_b128(
        (gv4i_p)(uintptr_t)(ksrc + (size_t)i * 8),
        (lv4i_p)(uint32_t)(uintptr_t)(Kd + (size_t)i * 8), 0, 0);
  }
#else
  for (int i = tid; i < 2048; i += 256) {
    ((uint4*)Qd)[i] = ((const uint4*)qsrc)[i];
    ((uint4*)Kd)[i] = ((const uint4*)ksrc)[i];
  }
#endif
#if HAS_TDM
  if (tid == 0) {  // TDM: EXEC-independent, one issue from wave 0
    unsigned long long ga = (unsigned long long)(uintptr_t)vsrc;
    u32x4 g0;
    g0[0] = 1u;                                  // count=1, no gather/restore
    g0[1] = (unsigned int)(uintptr_t)Vd;         // lds_addr (low 32 bits = LDS byte addr)
    g0[2] = (unsigned int)(ga & 0xFFFFFFFFull);  // global_addr[31:0]
    g0[3] = (unsigned int)((ga >> 32) & 0x1FFFFFFull) | (2u << 30);  // addr hi | type=2
    i32x8 g1;
    g1[0] = (int)(1u << 16);      // workgroup_mask=0, data_size=1 (2 bytes)
    g1[1] = (int)(32u << 16);     // tensor_dim0 = 32 elems (bits 79:48, lo half)
    g1[2] = (int)(32768u << 16);  // tensor_dim0 hi=0 | tensor_dim1 lo = 32768
    g1[3] = (int)(32u << 16);     // tensor_dim1 hi=0 | tile_dim0 = 32
    g1[4] = 128;                  // tile_dim1 = 128, tile_dim2 = 0
    g1[5] = 128;                  // tensor_dim0_stride = 128 elems (row pitch)
    g1[6] = 0;
    g1[7] = 0;
    i32x4 gz = {0, 0, 0, 0};
#if __clang_major__ >= 23
    i32x8 gz8 = {0, 0, 0, 0, 0, 0, 0, 0};
    __builtin_amdgcn_tensor_load_to_lds(g0, g1, gz, gz, gz8, 0);
#else
    __builtin_amdgcn_tensor_load_to_lds(g0, g1, gz, gz, 0);
#endif
  }
#elif HAS_ASYNC_LDS
  for (int i = tid; i < 512; i += 256) {
    int r = i >> 2, pp = i & 3;
    __builtin_amdgcn_global_load_async_to_lds_b128(
        (gv4i_p)(uintptr_t)(vsrc + (size_t)r * 128 + pp * 8),
        (lv4i_p)(uint32_t)(uintptr_t)(Vd + (size_t)r * 32 + pp * 8), 0, 0);
  }
#else
  for (int i = tid; i < 512; i += 256) {
    int r = i >> 2, pp = i & 3;
    ((uint4*)Vd)[r * 4 + pp] = *(const uint4*)(vsrc + (size_t)r * 128 + pp * 8);
  }
#endif
}

__device__ __forceinline__ void wait_copies() {
#if HAS_ASYNC_LDS
#if __has_builtin(__builtin_amdgcn_s_wait_asynccnt)
  __builtin_amdgcn_s_wait_asynccnt(0);
#endif
#endif
#if HAS_TDM
#if __has_builtin(__builtin_amdgcn_s_wait_tensorcnt)
  __builtin_amdgcn_s_wait_tensorcnt(0);
#endif
#endif
}

// ---------------------------------------------------------------------------
// K0: convert weights to f16 (fold dn into proj; pad proj to 112 rows)
// ---------------------------------------------------------------------------
__global__ void wcvt_kernel(const float* wq, const float* wk, const float* wv,
                            const float* proj, const float* wo, const float* f1,
                            const float* f2,
                            _Float16* wqh, _Float16* wkh, _Float16* wvh,
                            _Float16* projh, _Float16* woh, _Float16* f1h,
                            _Float16* f2h) {
  int tid = blockIdx.x * 256 + threadIdx.x;
  int str = gridDim.x * 256;
  for (int i = tid; i < 4096; i += str) {
    wqh[i] = (_Float16)wq[i];
    wkh[i] = (_Float16)wk[i];
    wvh[i] = (_Float16)wv[i];
    woh[i] = (_Float16)wo[i];
    f1h[i] = (_Float16)f1[i];
    f2h[i] = (_Float16)f2[i];
  }
  const float dn = 0.4204482076268573f;  // 32^-0.25
  for (int i = tid; i < 112 * 32; i += str) {
    int f = i >> 5;
    projh[i] = (_Float16)(f < 110 ? proj[i] * dn : 0.f);
  }
}

// ---------------------------------------------------------------------------
// K1: per-voxel 3->32 linear + LayerNorm1 -> tok (f32), h1 (f16)
// ---------------------------------------------------------------------------
__global__ __launch_bounds__(256) void preproc_kernel(
    const float* __restrict__ x, const float* __restrict__ pw,
    const float* __restrict__ pb, const float* __restrict__ g1,
    const float* __restrict__ b1, float* __restrict__ tok,
    _Float16* __restrict__ h1) {
  size_t i = (size_t)blockIdx.x * 256 + threadIdx.x;  // token id, 0..262143
  int b = (int)(i >> 15);
  int r = (int)(i & 32767);
  const float* xb = x + (size_t)b * 3 * 32768 + r;
  float x0 = xb[0], x1 = xb[32768], x2 = xb[65536];
  float vch[32];
  float m = 0.f;
#pragma unroll
  for (int o = 0; o < 32; ++o) {
    float t = pw[o * 3] * x0 + pw[o * 3 + 1] * x1 + pw[o * 3 + 2] * x2 + pb[o];
    vch[o] = t;
    m += t;
  }
  m *= (1.f / 32.f);
  float var = 0.f;
#pragma unroll
  for (int o = 0; o < 32; ++o) {
    float d = vch[o] - m;
    var += d * d;
  }
  var *= (1.f / 32.f);
  float rs = rsqrtf(var + 1e-5f);
  float* trow = tok + i * 32;
  _Float16* hrow = h1 + i * 32;
#pragma unroll
  for (int o = 0; o < 32; ++o) {
    trow[o] = vch[o];
    hrow[o] = (_Float16)((vch[o] - m) * rs * g1[o] + b1[o]);
  }
}

// ---------------------------------------------------------------------------
// K2: QKV projection. One wave per 16-token tile. 24 WMMAs/wave (K=32).
// ---------------------------------------------------------------------------
__global__ __launch_bounds__(256) void qkv_kernel(
    const _Float16* __restrict__ h1, const _Float16* __restrict__ wqh,
    const _Float16* __restrict__ wkh, const _Float16* __restrict__ wvh,
    const float* __restrict__ bq, const float* __restrict__ bk,
    const float* __restrict__ bv, _Float16* __restrict__ q,
    _Float16* __restrict__ k, _Float16* __restrict__ v) {
  int wid = blockIdx.x * 8 + (threadIdx.x >> 5);
  size_t row0 = (size_t)wid * 16;
  int lane = threadIdx.x & 31;
  int nn = lane & 15, mo = (lane & 16) ? 8 : 0;
  v16h a = load_a_f16(h1 + row0 * 32, 32);
  const _Float16* Ws[3] = {wqh, wkh, wvh};
  const float* Bs[3] = {bq, bk, bv};
  _Float16* Os[3] = {q, k, v};
#pragma unroll
  for (int m3 = 0; m3 < 3; ++m3) {
#pragma unroll
    for (int nt = 0; nt < 8; ++nt) {
      v16h bt = load_b_nk_f16(Ws[m3] + (size_t)(nt * 16) * 32, 32);
      v8f c = {};
      c = wmma_f(a, bt, c);
      float bias = Bs[m3][nt * 16 + nn];
#pragma unroll
      for (int r = 0; r < 8; ++r)
        Os[m3][(row0 + r + mo) * 128 + nt * 16 + nn] = (_Float16)(c[r] + bias);
    }
  }
}

// ---------------------------------------------------------------------------
// K3: FAVOR+ softmax features.
// ---------------------------------------------------------------------------
__global__ __launch_bounds__(256) void feat_kernel(
    const _Float16* __restrict__ q, const _Float16* __restrict__ k,
    const _Float16* __restrict__ projh, _Float16* __restrict__ qp,
    _Float16* __restrict__ kp) {
  int wid = blockIdx.x * 8 + (threadIdx.x >> 5);
  int lane = threadIdx.x & 31;
  int tt = wid >> 3;
  int rem = wid & 7;
  int h = rem >> 1;
  int isq = rem & 1;
  size_t row0 = (size_t)tt * 16;
  int b = (int)(row0 >> 15);
  int tin = (int)(row0 & 32767);
  const _Float16* src = (isq ? q : k) + row0 * 128 + h * 32;
  _Float16* dst = (isq ? qp : kp) + ((size_t)(b * 4 + h) * 32768 + tin) * 128;

  v16h a = load_a_f16(src, 128);
  float local = 0.f;
#pragma unroll
  for (int j = 0; j < 16; ++j) {
    float t = (float)a[j];
    local += t * t;
  }
  float full = local + __shfl_xor(local, 16, 32);  // full row sumsq (row lane&15)
  const float dn2 = 0.1767766952966369f;           // 32^-0.5
  float diag = 0.5f * dn2 * full;

  v8f dd[7];
#pragma unroll
  for (int ft = 0; ft < 7; ++ft) {
    v16h bt = load_b_nk_f16(projh + (size_t)(ft * 16) * 32, 32);
    v8f z = {};
    dd[ft] = wmma_f(a, bt, z);
  }
  int nn = lane & 15, mo = (lane & 16) ? 8 : 0;
  float rm[8];
#pragma unroll
  for (int r = 0; r < 8; ++r) {
    float xm = -1e30f;
#pragma unroll
    for (int ft = 0; ft < 7; ++ft) xm = fmaxf(xm, dd[ft][r]);
#pragma unroll
    for (int mk = 1; mk <= 8; mk <<= 1) xm = fmaxf(xm, __shfl_xor(xm, mk, 32));
    rm[r] = xm;
  }
  const float ratio = 0.09534625892455924f;  // 110^-0.5
#pragma unroll
  for (int r = 0; r < 8; ++r) {
    float dg = __shfl(diag, r + mo, 32);
#pragma unroll
    for (int ft = 0; ft < 7; ++ft) {
      float val = ratio * (expf(dd[ft][r] - dg - rm[r]) + 1e-4f);
      if (ft == 6 && nn >= 14) val = 0.f;  // features 110,111 are padding
      dst[(size_t)(r + mo) * 128 + ft * 16 + nn] = (_Float16)val;
    }
    dst[(size_t)(r + mo) * 128 + 112 + nn] = (_Float16)0.f;
  }
}

// ---------------------------------------------------------------------------
// K4: chunked causal linear attention, double-buffered with async/TDM staging.
// One workgroup (8 waves) per (b,h); 256 chunks of 128 tokens. Per chunk:
//   A = tril(Qc Kc^T); z = rowsum(A) + Qc.ksum
//   out = (A @ Vc + Qc @ Sh) / z
//   S += Kc^T @ Vc ; ksum += colsum(Kc)
// while the next chunk streams into the other LDS buffer.
// ---------------------------------------------------------------------------
__global__ __launch_bounds__(256) void attn_kernel(
    const _Float16* __restrict__ qp, const _Float16* __restrict__ kp,
    const _Float16* __restrict__ v, _Float16* __restrict__ attn_out) {
  __shared__ _Float16 Qc[2][128 * 128];
  __shared__ _Float16 Kc[2][128 * 128];
  __shared__ _Float16 Vc[2][128 * 32];
  __shared__ _Float16 Ah[128 * 128];
  __shared__ _Float16 Sh[128 * 32];
  __shared__ float Sf[128 * 32];
  __shared__ float ksum[128];
  __shared__ float zrow[128];

  int bh = blockIdx.x;
  int b = bh >> 2, h = bh & 3;
  int tid = threadIdx.x;
  int lane = tid & 31;
  int w = tid >> 5;
  int nn = lane & 15, mo = (lane & 16) ? 8 : 0;

  for (int i = tid; i < 128 * 32; i += 256) Sf[i] = 0.f;
  for (int i = tid; i < 128; i += 256) ksum[i] = 0.f;

  const _Float16* qpb = qp + (size_t)bh * 32768 * 128;
  const _Float16* kpb = kp + (size_t)bh * 32768 * 128;
  const _Float16* vb = v + (size_t)b * 32768 * 128 + h * 32;
  _Float16* ob = attn_out + (size_t)b * 32768 * 128 + h * 32;

  // preload chunk 0 into buffer 0
  copy_chunk(qpb, kpb, vb, Qc[0], Kc[0], Vc[0], tid);
  wait_copies();
  __syncthreads();

  for (int c = 0; c < 256; ++c) {
    int bb = c & 1;
    const _Float16* Q = Qc[bb];
    const _Float16* K = Kc[bb];
    const _Float16* V = Vc[bb];

    // ---- stage chunk c+1 into the other buffer (async / TDM) ----
    if (c + 1 < 256) {
      copy_chunk(qpb + (size_t)(c + 1) * 128 * 128,
                 kpb + (size_t)(c + 1) * 128 * 128,
                 vb + (size_t)(c + 1) * 128 * 128, Qc[bb ^ 1], Kc[bb ^ 1],
                 Vc[bb ^ 1], tid);
    }
    if (c + 2 < 256) {  // L2 prefetch two chunks ahead -> global_prefetch_b8
      __builtin_prefetch(qpb + (size_t)(c + 2) * 128 * 128 + tid * 64, 0, 1);
      __builtin_prefetch(kpb + (size_t)(c + 2) * 128 * 128 + tid * 64, 0, 1);
    }
    // state -> f16 for WMMA B operand
    for (int i = tid; i < 128 * 32; i += 256) Sh[i] = (_Float16)Sf[i];
    __syncthreads();

    // ---- z_inter = Qc . ksum (+eps * rowsum(Qc)) ----
    if (tid < 128) {
      float acc = 0.f, qs = 0.f;
      for (int f = 0; f < 128; ++f) {
        float qv = (float)Q[tid * 128 + f];
        acc += qv * ksum[f];
        qs += qv;
      }
      zrow[tid] = acc + 1e-6f * qs;
    }
    __syncthreads();

    // ---- Phase A: masked A = tril(Qc Kc^T), M-tile w ----
    {
      float rs[8];
#pragma unroll
      for (int r = 0; r < 8; ++r) rs[r] = 0.f;
      for (int nt = 0; nt <= w; ++nt) {
        v8f acc = {};
#pragma unroll
        for (int ks = 0; ks < 4; ++ks) {
          v16h a = load_a_f16(Q + (size_t)(w * 16) * 128 + ks * 32, 128);
          v16h bt = load_b_nk_f16(K + (size_t)(nt * 16) * 128 + ks * 32, 128);
          acc = wmma_f(a, bt, acc);
        }
#pragma unroll
        for (int r = 0; r < 8; ++r) {
          int m = r + mo;
          float val = acc[r];
          if (nt == w && nn > m) val = 0.f;  // causal mask in diagonal tile
          rs[r] += val;
          Ah[(size_t)(w * 16 + m) * 128 + nt * 16 + nn] = (_Float16)val;
        }
      }
      for (int nt = w + 1; nt < 8; ++nt) {
#pragma unroll
        for (int r = 0; r < 8; ++r)
          Ah[(size_t)(w * 16 + r + mo) * 128 + nt * 16 + nn] = (_Float16)0.f;
      }
#pragma unroll
      for (int r = 0; r < 8; ++r) {
        float xsum = rs[r];
#pragma unroll
        for (int mk = 1; mk <= 8; mk <<= 1) xsum += __shfl_xor(xsum, mk, 32);
        rs[r] = xsum;
      }
      if (nn == 0) {  // lanes 0 and 16 write rows 0..7 / 8..15
#pragma unroll
        for (int r = 0; r < 8; ++r) zrow[w * 16 + r + mo] += rs[r];
      }
    }
    __syncthreads();

    // ---- Phase B: out = (Ah @ Vc + Qc @ Sh) / z ----
    {
      v8f o0 = {}, o1 = {};
#pragma unroll
      for (int ks = 0; ks < 4; ++ks) {
        v16h a = load_a_f16(Ah + (size_t)(w * 16) * 128 + ks * 32, 128);
        v16h b0 = load_b_kn_f16(V + (size_t)(ks * 32) * 32 + 0, 32);
        v16h b1 = load_b_kn_f16(V + (size_t)(ks * 32) * 32 + 16, 32);
        o0 = wmma_f(a, b0, o0);
        o1 = wmma_f(a, b1, o1);
      }
#pragma unroll
      for (int ks = 0; ks < 4; ++ks) {
        v16h a = load_a_f16(Q + (size_t)(w * 16) * 128 + ks * 32, 128);
        v16h b0 = load_b_kn_f16(Sh + (size_t)(ks * 32) * 32 + 0, 32);
        v16h b1 = load_b_kn_f16(Sh + (size_t)(ks * 32) * 32 + 16, 32);
        o0 = wmma_f(a, b0, o0);
        o1 = wmma_f(a, b1, o1);
      }
#pragma unroll
      for (int r = 0; r < 8; ++r) {
        int m = w * 16 + r + mo;
        float inv = 1.f / (zrow[m] + 1e-6f);
        size_t row = (size_t)(c * 128 + m) * 128;
        ob[row + nn] = (_Float16)(o0[r] * inv);
        ob[row + 16 + nn] = (_Float16)(o1[r] * inv);
      }
    }
    __syncthreads();

    // ---- Phase C: S += Kc^T @ Vc ; ksum += colsum(Kc) ----
    {
      v8f s0, s1;
#pragma unroll
      for (int r = 0; r < 8; ++r) {
        s0[r] = Sf[(size_t)(w * 16 + r + mo) * 32 + nn];
        s1[r] = Sf[(size_t)(w * 16 + r + mo) * 32 + 16 + nn];
      }
      int fr = w * 16 + (lane & 15);
      int kb = (lane & 16) ? 16 : 0;
#pragma unroll
      for (int ks = 0; ks < 4; ++ks) {
        v16h a;  // A = Kc^T tile: row m = feature fr, K = token
#pragma unroll
        for (int j = 0; j < 8; ++j)
          a[j] = K[(size_t)(ks * 32 + kb + j) * 128 + fr];
#pragma unroll
        for (int j = 0; j < 8; ++j)
          a[8 + j] = K[(size_t)(ks * 32 + kb + 8 + j) * 128 + fr];
        v16h b0 = load_b_kn_f16(V + (size_t)(ks * 32) * 32 + 0, 32);
        v16h b1 = load_b_kn_f16(V + (size_t)(ks * 32) * 32 + 16, 32);
        s0 = wmma_f(a, b0, s0);
        s1 = wmma_f(a, b1, s1);
      }
#pragma unroll
      for (int r = 0; r < 8; ++r) {
        Sf[(size_t)(w * 16 + r + mo) * 32 + nn] = s0[r];
        Sf[(size_t)(w * 16 + r + mo) * 32 + 16 + nn] = s1[r];
      }
      if (tid < 128) {
        float acc = 0.f;
        for (int t = 0; t < 128; ++t) acc += (float)K[t * 128 + tid];
        ksum[tid] += acc;
      }
    }
    __syncthreads();
    // ---- ensure next chunk's DMA has landed before reusing buffers ----
    wait_copies();
    __syncthreads();
  }
}

// ---------------------------------------------------------------------------
// K5: out32 = residual + in128 @ W[32][128]^T + bias   (used for wo and ff2)
// ---------------------------------------------------------------------------
__global__ __launch_bounds__(256) void proj32_kernel(
    const _Float16* __restrict__ in128, const _Float16* __restrict__ wh,
    const float* __restrict__ bias, const float* __restrict__ resid,
    float* __restrict__ out32) {
  int wid = blockIdx.x * 8 + (threadIdx.x >> 5);
  size_t row0 = (size_t)wid * 16;
  int lane = threadIdx.x & 31;
  int nn = lane & 15, mo = (lane & 16) ? 8 : 0;
#pragma unroll
  for (int nt = 0; nt < 2; ++nt) {
    v8f c = {};
#pragma unroll
    for (int ks = 0; ks < 4; ++ks) {
      v16h a = load_a_f16(in128 + row0 * 128 + ks * 32, 128);
      v16h bt = load_b_nk_f16(wh + (size_t)(nt * 16) * 128 + ks * 32, 128);
      c = wmma_f(a, bt, c);
    }
    float bval = bias[nt * 16 + nn];
#pragma unroll
    for (int r = 0; r < 8; ++r) {
      size_t idx = (row0 + r + mo) * 32 + nt * 16 + nn;
      out32[idx] = resid[idx] + c[r] + bval;
    }
  }
}

// ---------------------------------------------------------------------------
// K6: LayerNorm over 32 channels -> f16
// ---------------------------------------------------------------------------
__global__ __launch_bounds__(256) void ln_kernel(const float* __restrict__ src,
                                                 const float* __restrict__ g,
                                                 const float* __restrict__ bta,
                                                 _Float16* __restrict__ dst) {
  size_t i = (size_t)blockIdx.x * 256 + threadIdx.x;
  const float* row = src + i * 32;
  float m = 0.f;
#pragma unroll
  for (int cc = 0; cc < 32; ++cc) m += row[cc];
  m *= (1.f / 32.f);
  float var = 0.f;
#pragma unroll
  for (int cc = 0; cc < 32; ++cc) {
    float d = row[cc] - m;
    var += d * d;
  }
  var *= (1.f / 32.f);
  float rs = rsqrtf(var + 1e-5f);
#pragma unroll
  for (int cc = 0; cc < 32; ++cc)
    dst[i * 32 + cc] = (_Float16)((row[cc] - m) * rs * g[cc] + bta[cc]);
}

// ---------------------------------------------------------------------------
// K7: FF1: gelu(h2 @ W1^T + b1) -> f16 [NT,128]
// ---------------------------------------------------------------------------
__global__ __launch_bounds__(256) void ff1_kernel(
    const _Float16* __restrict__ h2, const _Float16* __restrict__ f1h,
    const float* __restrict__ b1, _Float16* __restrict__ ff1) {
  int wid = blockIdx.x * 8 + (threadIdx.x >> 5);
  size_t row0 = (size_t)wid * 16;
  int lane = threadIdx.x & 31;
  int nn = lane & 15, mo = (lane & 16) ? 8 : 0;
  v16h a = load_a_f16(h2 + row0 * 32, 32);
#pragma unroll
  for (int nt = 0; nt < 8; ++nt) {
    v16h bt = load_b_nk_f16(f1h + (size_t)(nt * 16) * 32, 32);
    v8f c = {};
    c = wmma_f(a, bt, c);
    float bval = b1[nt * 16 + nn];
#pragma unroll
    for (int r = 0; r < 8; ++r) {
      float val = c[r] + bval;
      float gl = 0.5f * val * (1.f + erff(val * 0.7071067811865475f));
      ff1[(row0 + r + mo) * 128 + nt * 16 + nn] = (_Float16)gl;
    }
  }
}

// ---------------------------------------------------------------------------
// K8: readout: t = T-1 slice, concat x[:,2,0], 5x5 SAME conv (33->1),
// global mean, final 1x1 linear. One block per batch.
// ---------------------------------------------------------------------------
__global__ __launch_bounds__(256) void readout_kernel(
    const float* __restrict__ tok, const float* __restrict__ x,
    const float* __restrict__ tcw, const float* __restrict__ tcb,
    const float* __restrict__ row_, const float* __restrict__ rob,
    float* __restrict__ out) {
  __shared__ float red[256];
  int b = blockIdx.x;
  int tid = threadIdx.x;
  float acc = 0.f;
  for (int p = tid; p < 1024; p += 256) {
    int hh = p >> 5, ww = p & 31;
    float s = 0.f;
    for (int ci = 0; ci < 33; ++ci) {
      for (int ky = 0; ky < 5; ++ky) {
        int yy = hh + ky - 2;
        if (yy < 0 || yy > 31) continue;
        for (int kx = 0; kx < 5; ++kx) {
          int xx = ww + kx - 2;
          if (xx < 0 || xx > 31) continue;
          float inp;
          if (ci < 32) {
            size_t row = (size_t)b * 32768 + 31 * 1024 + yy * 32 + xx;
            inp = tok[row * 32 + ci];
          } else {
            inp = x[((size_t)b * 3 + 2) * 32768 + yy * 32 + xx];
          }
          s += inp * tcw[(ci * 5 + ky) * 5 + kx];
        }
      }
    }
    acc += s;
  }
  red[tid] = acc;
  __syncthreads();
  for (int st = 128; st > 0; st >>= 1) {
    if (tid < st) red[tid] += red[tid + st];
    __syncthreads();
  }
  if (tid == 0) {
    float mean = red[0] * (1.f / 1024.f) + tcb[0];
    out[b] = mean * row_[0] + rob[0];
  }
}

// ---------------------------------------------------------------------------
extern "C" void kernel_launch(void* const* d_in, const int* in_sizes, int n_in,
                              void* d_out, int out_size, void* d_ws,
                              size_t ws_size, hipStream_t stream) {
  (void)in_sizes; (void)n_in; (void)out_size; (void)ws_size;
  const float* x    = (const float*)d_in[0];
  const float* pw   = (const float*)d_in[1];
  const float* pb   = (const float*)d_in[2];
  const float* g1   = (const float*)d_in[3];
  const float* b1v  = (const float*)d_in[4];
  const float* wq   = (const float*)d_in[5];
  const float* bq   = (const float*)d_in[6];
  const float* wk   = (const float*)d_in[7];
  const float* bk   = (const float*)d_in[8];
  const float* wv   = (const float*)d_in[9];
  const float* bv   = (const float*)d_in[10];
  const float* proj = (const float*)d_in[11];
  const float* wo   = (const float*)d_in[12];
  const float* bo   = (const float*)d_in[13];
  const float* g2   = (const float*)d_in[14];
  const float* b2v  = (const float*)d_in[15];
  const float* fw1  = (const float*)d_in[16];
  const float* fb1  = (const float*)d_in[17];
  const float* fw2  = (const float*)d_in[18];
  const float* fb2  = (const float*)d_in[19];
  const float* tcw  = (const float*)d_in[20];
  const float* tcb  = (const float*)d_in[21];
  const float* row  = (const float*)d_in[22];
  const float* rob  = (const float*)d_in[23];
  float* out = (float*)d_out;

  const size_t NT = 262144;  // 8 * 32768 tokens
  char* p = (char*)d_ws;
  auto alloc = [&](size_t bytes) -> char* {
    char* r = p;
    p += (bytes + 255) & ~(size_t)255;
    return r;
  };
  float*    tok   = (float*)alloc(NT * 32 * 4);
  float*    tok2  = (float*)alloc(NT * 32 * 4);
  _Float16* h1    = (_Float16*)alloc(NT * 32 * 2);
  _Float16* h2    = (_Float16*)alloc(NT * 32 * 2);
  _Float16* qb    = (_Float16*)alloc(NT * 128 * 2);
  _Float16* kb2   = (_Float16*)alloc(NT * 128 * 2);
  _Float16* vb    = (_Float16*)alloc(NT * 128 * 2);
  _Float16* attn  = (_Float16*)alloc(NT * 128 * 2);
  _Float16* ff1   = (_Float16*)alloc(NT * 128 * 2);
  _Float16* qp    = (_Float16*)alloc((size_t)32 * 32768 * 128 * 2);
  _Float16* kp    = (_Float16*)alloc((size_t)32 * 32768 * 128 * 2);
  _Float16* wqh   = (_Float16*)alloc(4096 * 2);
  _Float16* wkh   = (_Float16*)alloc(4096 * 2);
  _Float16* wvh   = (_Float16*)alloc(4096 * 2);
  _Float16* projh = (_Float16*)alloc(112 * 32 * 2);
  _Float16* woh   = (_Float16*)alloc(4096 * 2);
  _Float16* f1h   = (_Float16*)alloc(4096 * 2);
  _Float16* f2h   = (_Float16*)alloc(4096 * 2);

  wcvt_kernel<<<16, 256, 0, stream>>>(wq, wk, wv, proj, wo, fw1, fw2, wqh, wkh,
                                      wvh, projh, woh, f1h, f2h);
  preproc_kernel<<<NT / 256, 256, 0, stream>>>(x, pw, pb, g1, b1v, tok, h1);
  qkv_kernel<<<NT / 128, 256, 0, stream>>>(h1, wqh, wkh, wvh, bq, bk, bv, qb,
                                           kb2, vb);
  feat_kernel<<<16384, 256, 0, stream>>>(qb, kb2, projh, qp, kp);
  attn_kernel<<<32, 256, 0, stream>>>(qp, kp, vb, attn);
  proj32_kernel<<<NT / 128, 256, 0, stream>>>(attn, woh, bo, tok, tok2);
  ln_kernel<<<NT / 256, 256, 0, stream>>>(tok2, g2, b2v, h2);
  ff1_kernel<<<NT / 128, 256, 0, stream>>>(h2, f1h, fb1, ff1);
  proj32_kernel<<<NT / 128, 256, 0, stream>>>(ff1, f2h, fb2, tok2, tok);
  readout_kernel<<<8, 256, 0, stream>>>(tok, x, tcw, tcb, row, rob, out);
}
